// TripleAttention_71116068487397
// MI455X (gfx1250) — matrix operand order repaired and checked by reading, not compiled, not run on hardware
//
#include <hip/hip_runtime.h>
#include <hip/hip_bf16.h>

// MI455X / gfx1250, wave32. bf16 WMMA (v_wmma_f32_16x16x32_bf16) GEMM pipeline.
// Fragment products built with native __bf16 vector ops -> v_pk_mul_bf16.

typedef __attribute__((ext_vector_type(16))) __bf16 v16bf;
typedef __attribute__((ext_vector_type(8)))  float  v8f;

#define B_ 4
#define N_ 4096
#define C_ 512
#define H_ 8
#define D_ 64

union BFrag {
  v16bf v;
  uint4 q[2];
  unsigned short h[16];
};

__device__ inline unsigned short f2bf(float x) {
  unsigned u = __builtin_bit_cast(unsigned, x);
  u += 0x7FFFu + ((u >> 16) & 1u);   // round-to-nearest-even
  return (unsigned short)(u >> 16);
}
__device__ inline float bf2f(unsigned short h) {
  unsigned u = (unsigned)h << 16;
  return __builtin_bit_cast(float, u);
}
__device__ inline v8f zero8() {
  v8f z;
#pragma unroll
  for (int e = 0; e < 8; ++e) z[e] = 0.f;
  return z;
}
__device__ inline v8f wmma_bf16(const BFrag& a, const BFrag& b, v8f c) {
  return __builtin_amdgcn_wmma_f32_16x16x32_bf16(false, a.v, false, b.v,
                                                 (short)0, c, false, false);
}

// ---------------- stage 0a: x (f32) -> xb (bf16) ----------------
__global__ __launch_bounds__(256) void k_cvt_x(const float* __restrict__ x,
                                               unsigned short* __restrict__ xb) {
  size_t t = (size_t)blockIdx.x * 256 + threadIdx.x;
  const float4* p = (const float4*)(x + t * 8);
  float4 a = p[0], b = p[1];
  union { unsigned short s[8]; uint4 q; } o;
  o.s[0] = f2bf(a.x); o.s[1] = f2bf(a.y); o.s[2] = f2bf(a.z); o.s[3] = f2bf(a.w);
  o.s[4] = f2bf(b.x); o.s[5] = f2bf(b.y); o.s[6] = f2bf(b.z); o.s[7] = f2bf(b.w);
  *(uint4*)(xb + t * 8) = o.q;
}

// ---------------- stage 0b: pack 6 weights into B-fragment layout ----------------
// wpack[w][kt][ntile][lane][e] = bf16(W[kt*32 + (e<8 ? (lane>>4)*8 + e : 16 + (lane>>4)*8 + e-8), ntile*16 + (lane&15)])
__global__ __launch_bounds__(256) void k_pack_w(const float* __restrict__ W0, const float* __restrict__ W1,
                                                const float* __restrict__ W2, const float* __restrict__ W3,
                                                const float* __restrict__ W4, const float* __restrict__ W5,
                                                unsigned short* __restrict__ wpack) {
  unsigned t = blockIdx.x * 256 + threadIdx.x;
  unsigned lane = t & 31, nt = (t >> 5) & 31, kt = (t >> 10) & 15, w = t >> 14;
  unsigned hi = lane >> 4, lr = lane & 15;
  const float* W = w == 0 ? W0 : w == 1 ? W1 : w == 2 ? W2 : w == 3 ? W3 : w == 4 ? W4 : W5;
  unsigned n = nt * 16 + lr;
  union { unsigned short s[16]; uint4 q[2]; } o;
#pragma unroll
  for (int e = 0; e < 16; ++e) {
    unsigned k = kt * 32 + (e < 8 ? hi * 8 + e : 16 + hi * 8 + (e - 8));
    o.s[e] = f2bf(W[(size_t)k * C_ + n]);
  }
  uint4* dst = (uint4*)(wpack + (((size_t)(w * 16 + kt) * 32 + nt) * 32 + lane) * 16);
  dst[0] = o.q[0];
  dst[1] = o.q[1];
}

// ---------------- stage 1: 5 projections  y = x + x@W + b, heads split ----------------
// p<2 (q1,q2): store (bh, n, d) bf16.  p>=2 (k1,k2,v): store transposed (bh, d, n) * 1/16.
__global__ __launch_bounds__(128) void k_proj(
    const float* __restrict__ x, const unsigned short* __restrict__ xb,
    const unsigned short* __restrict__ wpack,
    const float* __restrict__ bq1, const float* __restrict__ bq2,
    const float* __restrict__ bk1, const float* __restrict__ bk2,
    const float* __restrict__ bv,
    unsigned short* __restrict__ q1, unsigned short* __restrict__ q2,
    unsigned short* __restrict__ k1t, unsigned short* __restrict__ k2t,
    unsigned short* __restrict__ vt) {
  const unsigned p = blockIdx.z;
  const float* bias = p == 0 ? bq1 : p == 1 ? bq2 : p == 2 ? bk1 : p == 3 ? bk2 : bv;
  unsigned short* dst = p == 0 ? q1 : p == 1 ? q2 : p == 2 ? k1t : p == 3 ? k2t : vt;
  const bool tl = p >= 2;
  const float scale = tl ? 0.0625f : 1.0f;  // N^(-1/3) = 4096^(-1/3) = 1/16
  const unsigned short* wp0 = wpack + (size_t)p * 262144;

  const unsigned lane = threadIdx.x & 31, wave = threadIdx.x >> 5;
  const unsigned hi = lane >> 4, lr = lane & 15;
  const unsigned m0 = blockIdx.x * 128 + wave * 32;
  const unsigned n0 = blockIdx.y * 64;

  v8f acc[2][4];
#pragma unroll
  for (int i = 0; i < 2; ++i)
#pragma unroll
    for (int j = 0; j < 4; ++j) acc[i][j] = zero8();

  for (unsigned kt = 0; kt < 16; ++kt) {
    const unsigned k0 = kt * 32;
    BFrag a[2];
#pragma unroll
    for (int mt = 0; mt < 2; ++mt) {
      const unsigned short* r = xb + (size_t)(m0 + mt * 16 + lr) * C_ + k0 + hi * 8;
      a[mt].q[0] = *(const uint4*)r;
      a[mt].q[1] = *(const uint4*)(r + 16);
    }
#pragma unroll
    for (int nt = 0; nt < 4; ++nt) {
      BFrag b;
      const unsigned short* wp = wp0 + (((size_t)kt * 32 + (blockIdx.y * 4 + nt)) * 32 + lane) * 16;
      b.q[0] = *(const uint4*)wp;
      b.q[1] = *(const uint4*)(wp + 8);
      acc[0][nt] = wmma_bf16(a[0], b, acc[0][nt]);
      acc[1][nt] = wmma_bf16(a[1], b, acc[1][nt]);
    }
  }

#pragma unroll
  for (int mt = 0; mt < 2; ++mt) {
#pragma unroll
    for (int nt = 0; nt < 4; ++nt) {
      const unsigned c = n0 + nt * 16 + lr;
      const unsigned hd = c >> 6, d = c & 63;
      const float bs = bias[c];
      const unsigned ms = m0 + mt * 16 + hi * 8;   // 8 consecutive token rows
      const unsigned bh = (ms >> 12) * H_ + hd;
      if (tl) {
        union { unsigned short s[8]; uint4 q; } o;
#pragma unroll
        for (int v = 0; v < 8; ++v) {
          const unsigned m = ms + v;
          float val = acc[mt][nt][v] + bs + x[(size_t)m * C_ + c];
          o.s[v] = f2bf(val * scale);
        }
        *(uint4*)(dst + ((size_t)bh * 64 + d) * N_ + (ms & (N_ - 1))) = o.q;
      } else {
#pragma unroll
        for (int v = 0; v < 8; ++v) {
          const unsigned m = ms + v;
          float val = acc[mt][nt][v] + bs + x[(size_t)m * C_ + c];
          dst[((size_t)bh * N_ + (m & (N_ - 1))) * 64 + d] = f2bf(val);
        }
      }
    }
  }
}

// ---------------- stage 2: stateT[j,(ik)] = sum_n v^T[j,n] * (k1[n,i]*k2[n,k]) ----------------
__global__ __launch_bounds__(128) void k_state(
    const unsigned short* __restrict__ k1t, const unsigned short* __restrict__ k2t,
    const unsigned short* __restrict__ vt, unsigned short* __restrict__ stateT) {
  const unsigned bh = blockIdx.y;
  const unsigned lane = threadIdx.x & 31, wave = threadIdx.x >> 5;
  const unsigned hi = lane >> 4, lr = lane & 15;
  const unsigned ik0 = blockIdx.x * 64 + wave * 16;   // this wave's 16 (i,k) columns
  const unsigned i = ik0 >> 6;
  const unsigned kc = (ik0 & 63) + lr;
  const unsigned short* k1r = k1t + ((size_t)bh * 64 + i) * N_;
  const unsigned short* k2r = k2t + ((size_t)bh * 64 + kc) * N_;

  v8f acc[4];
#pragma unroll
  for (int t = 0; t < 4; ++t) acc[t] = zero8();

  for (unsigned n0 = 0; n0 < N_; n0 += 32) {
    // Build B fragment: B[n,(ik)] = k1[n,i] * k2[n,k]; 16 consecutive n per lane.
    // Native bf16 vector multiply -> v_pk_mul_bf16 x8.
    BFrag f1, f2, b;
    const unsigned short* p1 = k1r + n0 + hi * 16;
    f1.q[0] = *(const uint4*)p1;
    f1.q[1] = *(const uint4*)(p1 + 8);
    const unsigned short* p2 = k2r + n0 + hi * 16;
    f2.q[0] = *(const uint4*)p2;
    f2.q[1] = *(const uint4*)(p2 + 8);
    b.v = f1.v * f2.v;
#pragma unroll
    for (int t = 0; t < 4; ++t) {   // 4 j-tiles (j = 0..63)
      BFrag a;                       // A[j,n] = v^T[j,n]
      const unsigned short* vr = vt + ((size_t)bh * 64 + t * 16 + lr) * N_ + n0 + hi * 16;
      a.q[0] = *(const uint4*)vr;
      a.q[1] = *(const uint4*)(vr + 8);
      acc[t] = wmma_bf16(a, b, acc[t]);
    }
  }
#pragma unroll
  for (int t = 0; t < 4; ++t)
#pragma unroll
    for (int v = 0; v < 8; ++v) {
      const unsigned j = t * 16 + hi * 8 + v;
      stateT[((size_t)bh * 64 + j) * 4096 + ik0 + lr] = f2bf(acc[t][v]);
    }
}

// ---------------- stage 3: out[n,j] = sum_(ik) (q1[n,i]*q2[n,k]) * stateT[j,(ik)] ----------------
__global__ __launch_bounds__(128) void k_out(
    const unsigned short* __restrict__ q1, const unsigned short* __restrict__ q2,
    const unsigned short* __restrict__ stateT, float* __restrict__ attn) {
  const unsigned bh = blockIdx.y;
  const unsigned lane = threadIdx.x & 31, wave = threadIdx.x >> 5;
  const unsigned hi = lane >> 4, lr = lane & 15;
  const unsigned nt0 = blockIdx.x * 64 + wave * 16;
  const unsigned nr = nt0 + lr;
  const unsigned short* q1r = q1 + ((size_t)bh * N_ + nr) * 64;
  const unsigned short* q2r = q2 + ((size_t)bh * N_ + nr) * 64;

  v8f acc[4];
#pragma unroll
  for (int t = 0; t < 4; ++t) acc[t] = zero8();

  for (unsigned ik0 = 0; ik0 < 4096; ik0 += 32) {
    const unsigned i = ik0 >> 6;
    const unsigned kb = ik0 & 63;           // 0 or 32
    const __bf16 q1b = __builtin_bit_cast(__bf16, q1r[i]);
    // A fragment: A[n,(ik)] = q1[n,i] * q2[n,k]; broadcast-scalar bf16 multiply.
    BFrag qf, a;
    const unsigned short* p = q2r + kb + hi * 16;
    qf.q[0] = *(const uint4*)p;
    qf.q[1] = *(const uint4*)(p + 8);
    a.v = qf.v * q1b;
#pragma unroll
    for (int t = 0; t < 4; ++t) {   // 4 j-tiles
      BFrag b;
      const unsigned short* sp = stateT + ((size_t)bh * 64 + t * 16 + lr) * 4096 + ik0 + hi * 16;
      b.q[0] = *(const uint4*)sp;
      b.q[1] = *(const uint4*)(sp + 8);
      acc[t] = wmma_bf16(a, b, acc[t]);
    }
  }
  const unsigned b_ = bh >> 3, hd = bh & 7;
#pragma unroll
  for (int t = 0; t < 4; ++t)
#pragma unroll
    for (int v = 0; v < 8; ++v) {
      const unsigned n = nt0 + hi * 8 + v;
      const unsigned j = t * 16 + lr;
      attn[((size_t)b_ * N_ + n) * C_ + hd * 64 + j] = acc[t][v];
    }
}

// ---------------- stage 4a: LayerNorm over C=512, one wave32 per row ----------------
__global__ __launch_bounds__(256) void k_ln(const float* __restrict__ attn,
                                            const float* __restrict__ g,
                                            const float* __restrict__ bt,
                                            unsigned short* __restrict__ lnout) {
  const unsigned row = blockIdx.x * 8 + (threadIdx.x >> 5);
  const unsigned lane = threadIdx.x & 31;
  const float4* p4 = (const float4*)(attn + (size_t)row * C_ + lane * 16);
  float v[16];
#pragma unroll
  for (int q = 0; q < 4; ++q) {
    float4 f = p4[q];
    v[q * 4 + 0] = f.x; v[q * 4 + 1] = f.y; v[q * 4 + 2] = f.z; v[q * 4 + 3] = f.w;
  }
  float s = 0.f, ss = 0.f;
#pragma unroll
  for (int e = 0; e < 16; ++e) { s += v[e]; ss += v[e] * v[e]; }
#pragma unroll
  for (int off = 16; off > 0; off >>= 1) {
    s += __shfl_xor(s, off, 32);
    ss += __shfl_xor(ss, off, 32);
  }
  const float mu = s * (1.f / 512.f);
  const float var = ss * (1.f / 512.f) - mu * mu;
  const float rs = rsqrtf(var + 1e-5f);
  union { unsigned short s16[16]; uint4 q[2]; } o;
#pragma unroll
  for (int e = 0; e < 16; ++e) {
    const unsigned c = lane * 16 + e;
    o.s16[e] = f2bf((v[e] - mu) * rs * g[c] + bt[c]);
  }
  uint4* dp = (uint4*)(lnout + (size_t)row * C_ + lane * 16);
  dp[0] = o.q[0];
  dp[1] = o.q[1];
}

// ---------------- stage 4b: out = LN(out) @ Wo + bo  (f32 result) ----------------
__global__ __launch_bounds__(128) void k_final(
    const unsigned short* __restrict__ ln, const unsigned short* __restrict__ wpack,
    const float* __restrict__ bo, float* __restrict__ out) {
  const unsigned short* wp0 = wpack + (size_t)5 * 262144;  // Wo pack
  const unsigned lane = threadIdx.x & 31, wave = threadIdx.x >> 5;
  const unsigned hi = lane >> 4, lr = lane & 15;
  const unsigned m0 = blockIdx.x * 128 + wave * 32;
  const unsigned n0 = blockIdx.y * 64;

  v8f acc[2][4];
#pragma unroll
  for (int i = 0; i < 2; ++i)
#pragma unroll
    for (int j = 0; j < 4; ++j) acc[i][j] = zero8();

  for (unsigned kt = 0; kt < 16; ++kt) {
    const unsigned k0 = kt * 32;
    BFrag a[2];
#pragma unroll
    for (int mt = 0; mt < 2; ++mt) {
      const unsigned short* r = ln + (size_t)(m0 + mt * 16 + lr) * C_ + k0 + hi * 8;
      a[mt].q[0] = *(const uint4*)r;
      a[mt].q[1] = *(const uint4*)(r + 16);
    }
#pragma unroll
    for (int nt = 0; nt < 4; ++nt) {
      BFrag b;
      const unsigned short* wp = wp0 + (((size_t)kt * 32 + (blockIdx.y * 4 + nt)) * 32 + lane) * 16;
      b.q[0] = *(const uint4*)wp;
      b.q[1] = *(const uint4*)(wp + 8);
      acc[0][nt] = wmma_bf16(a[0], b, acc[0][nt]);
      acc[1][nt] = wmma_bf16(a[1], b, acc[1][nt]);
    }
  }

#pragma unroll
  for (int mt = 0; mt < 2; ++mt) {
#pragma unroll
    for (int nt = 0; nt < 4; ++nt) {
      const unsigned c = n0 + nt * 16 + lr;
      const float bs = bo[c];
      const unsigned ms = m0 + mt * 16 + hi * 8;
#pragma unroll
      for (int v = 0; v < 8; ++v) {
        out[(size_t)(ms + v) * C_ + c] = acc[mt][nt][v] + bs;
      }
    }
  }
}

extern "C" void kernel_launch(void* const* d_in, const int* in_sizes, int n_in,
                              void* d_out, int out_size, void* d_ws, size_t ws_size,
                              hipStream_t stream) {
  (void)in_sizes; (void)n_in; (void)out_size; (void)ws_size;
  const float* x   = (const float*)d_in[0];
  const float* Wq1 = (const float*)d_in[1];  const float* bq1 = (const float*)d_in[2];
  const float* Wq2 = (const float*)d_in[3];  const float* bq2 = (const float*)d_in[4];
  const float* Wk1 = (const float*)d_in[5];  const float* bk1 = (const float*)d_in[6];
  const float* Wk2 = (const float*)d_in[7];  const float* bk2 = (const float*)d_in[8];
  const float* Wv  = (const float*)d_in[9];  const float* bv  = (const float*)d_in[10];
  const float* lng = (const float*)d_in[11]; const float* lnb = (const float*)d_in[12];
  const float* Wo  = (const float*)d_in[13]; const float* bo  = (const float*)d_in[14];
  float* out = (float*)d_out;
  char* ws = (char*)d_ws;

  // Workspace layout (bytes), total ~115 MB with reuse:
  const size_t WPACK = 0;                         // 6 * 512KB packed bf16 weights
  const size_t XB    = WPACK + 6ull * 512 * 512 * 2;  // x bf16 (16MB); reused as LN output
  const size_t Q1    = XB + 16777216ull;          // q1 (bh,n,d) bf16
  const size_t Q2    = Q1 + 16777216ull;          // q2 (bh,n,d) bf16
  const size_t KVT   = Q2 + 16777216ull;          // k1t|k2t|vt (bh,d,n) bf16; reused as attn f32
  const size_t K1T = KVT, K2T = KVT + 16777216ull, VT = KVT + 33554432ull;
  const size_t STATE = KVT + 50331648ull;         // stateT (bh,j,ik) bf16 (16MB)

  unsigned short* xb  = (unsigned short*)(ws + XB);
  unsigned short* wpk = (unsigned short*)(ws + WPACK);
  unsigned short* q1  = (unsigned short*)(ws + Q1);
  unsigned short* q2  = (unsigned short*)(ws + Q2);
  unsigned short* k1t = (unsigned short*)(ws + K1T);
  unsigned short* k2t = (unsigned short*)(ws + K2T);
  unsigned short* vt  = (unsigned short*)(ws + VT);
  unsigned short* st  = (unsigned short*)(ws + STATE);
  float*          att = (float*)(ws + KVT);       // reuse (k-buffers dead after k_state)
  unsigned short* lno = (unsigned short*)(ws + XB); // reuse (xb dead after k_proj)

  k_cvt_x <<<4096, 256, 0, stream>>>(x, xb);
  k_pack_w<<<384, 256, 0, stream>>>(Wq1, Wq2, Wk1, Wk2, Wv, Wo, wpk);
  k_proj  <<<dim3(128, 8, 5), 128, 0, stream>>>(x, xb, wpk, bq1, bq2, bk1, bk2, bv,
                                                q1, q2, k1t, k2t, vt);
  k_state <<<dim3(64, 32), 128, 0, stream>>>(k1t, k2t, vt, st);
  k_out   <<<dim3(64, 32), 128, 0, stream>>>(q1, q2, st, att);
  k_ln    <<<2048, 256, 0, stream>>>(att, lng, lnb, lno);
  k_final <<<dim3(128, 8), 128, 0, stream>>>(lno, wpk, bo, out);
}
